// URLGNN_11776800326003
// MI455X (gfx1250) — matrix operand order repaired and checked by reading, not compiled, not run on hardware
//
#include <hip/hip_runtime.h>
#include <hip/hip_bf16.h>
#include <math.h>

typedef __attribute__((ext_vector_type(16))) _Float16 v16h;
typedef __attribute__((ext_vector_type(8)))  float    v8f;

// ---------------------------------------------------------------- utilities
__global__ void k_fill(float* __restrict__ p, float v, int n4) {
    int i = blockIdx.x * blockDim.x + threadIdx.x;   // n4 = n/4, n multiple of 4
    if (i < n4) {
        float4 q = make_float4(v, v, v, v);
        *(float4*)(p + (size_t)i * 4) = q;
    }
}

__global__ void k_degree(const int* __restrict__ dst, float* __restrict__ deg, int E) {
    int i = blockIdx.x * blockDim.x + threadIdx.x;
    if (i < E) atomicAdd(&deg[dst[i]], 1.0f);
}

__global__ void k_dis(const float* __restrict__ deg, float* __restrict__ dis, int N) {
    int i = blockIdx.x * blockDim.x + threadIdx.x;
    if (i < N) dis[i] = rsqrtf(fmaxf(deg[i], 1.0f));
}

// Pre-transpose weights to f16 [col][K] so each WMMA B-fragment is one
// contiguous 32-byte load per lane (no strided b32 loads, no runtime cvt).
__global__ void k_prepw(const float* __restrict__ W, _Float16* __restrict__ Wt,
                        int K, int C) {
    int i = blockIdx.x * blockDim.x + threadIdx.x;
    if (i >= K * C) return;
    int c = i / K, k = i - c * K;
    Wt[(size_t)c * K + k] = (_Float16)W[(size_t)k * C + c];
}

__device__ __forceinline__ void cvt8(v16h& a, const float4 f0, const float4 f1, int base) {
    a[base + 0] = (_Float16)f0.x; a[base + 1] = (_Float16)f0.y;
    a[base + 2] = (_Float16)f0.z; a[base + 3] = (_Float16)f0.w;
    a[base + 4] = (_Float16)f1.x; a[base + 5] = (_Float16)f1.y;
    a[base + 6] = (_Float16)f1.z; a[base + 7] = (_Float16)f1.w;
}

// ------------------------------------------------- GEMM1: emb-gather @ W1
// h1[N,128] = emb[node_ids][N,64] @ W1[64,128].  Block = 16-row tile, 8 waves
// = 8 column tiles.  A uses contiguous k = 16*half + s; B per-column, same order.
__global__ void __launch_bounds__(256) k_gemm1(const int* __restrict__ node_ids,
                                               const float* __restrict__ emb,
                                               const _Float16* __restrict__ W1t,
                                               float* __restrict__ h1, int N) {
    const int lane = threadIdx.x & 31;
    const int wv   = threadIdx.x >> 5;      // 0..7 -> output col tile
    const int tm   = blockIdx.x;            // row tile
    const int m    = lane & 15;
    const int hi   = lane >> 4;             // 0/1 half-wave
    int row = tm * 16 + m;
    if (row >= N) row = N - 1;              // clamp (N=100000 has no tail)
    const int col = wv * 16 + m;
    const float* __restrict__ er = emb + (size_t)node_ids[row] * 64;

    v8f c = {};
#pragma unroll
    for (int ks = 0; ks < 2; ++ks) {
        const int ka = ks * 32 + hi * 16;
        v16h a;
        cvt8(a, *(const float4*)(er + ka),      *(const float4*)(er + ka + 4),  0);
        cvt8(a, *(const float4*)(er + ka + 8), *(const float4*)(er + ka + 12), 8);
        const v16h b = *(const v16h*)(W1t + (size_t)col * 64 + ks * 32 + hi * 16);
        c = __builtin_amdgcn_wmma_f32_16x16x32_f16(false, a, false, b,
                                                   (short)0, c, false, false);
    }
    const bool full = (tm * 16 + 16 <= N);  // wave-uniform: one branch, no EXEC churn
    if (full) {
#pragma unroll
        for (int r = 0; r < 8; ++r)
            h1[(size_t)(tm * 16 + r + hi * 8) * 128 + col] = c[r];
    } else {
#pragma unroll
        for (int r = 0; r < 8; ++r) {
            int orow = tm * 16 + r + hi * 8;
            if (orow < N) h1[(size_t)orow * 128 + col] = c[r];
        }
    }
}

// ------------------------------------------------- GEMM2: x1 @ W2
// h2[N,64] = x1[N,128] @ W2[128,64].  Block = 2 row tiles x 4 col tiles.
__global__ void __launch_bounds__(256) k_gemm2(const float* __restrict__ x1,
                                               const _Float16* __restrict__ W2t,
                                               float* __restrict__ h2, int N) {
    const int lane  = threadIdx.x & 31;
    const int wv    = threadIdx.x >> 5;
    const int numTm = (N + 15) >> 4;
    const int tm    = blockIdx.x * 2 + (wv >> 2);
    const int tn    = wv & 3;
    if (tm >= numTm) return;                // wave-uniform
    const int m  = lane & 15;
    const int hi = lane >> 4;
    int row = tm * 16 + m;
    if (row >= N) row = N - 1;
    const int col = tn * 16 + m;
    const float* __restrict__ xr = x1 + (size_t)row * 128;

    v8f c = {};
#pragma unroll
    for (int ks = 0; ks < 4; ++ks) {
        const int ka = ks * 32 + hi * 16;
        v16h a;
        cvt8(a, *(const float4*)(xr + ka),      *(const float4*)(xr + ka + 4),  0);
        cvt8(a, *(const float4*)(xr + ka + 8), *(const float4*)(xr + ka + 12), 8);
        const v16h b = *(const v16h*)(W2t + (size_t)col * 128 + ks * 32 + hi * 16);
        c = __builtin_amdgcn_wmma_f32_16x16x32_f16(false, a, false, b,
                                                   (short)0, c, false, false);
    }
    const bool full = (tm * 16 + 16 <= N);
    if (full) {
#pragma unroll
        for (int r = 0; r < 8; ++r)
            h2[(size_t)(tm * 16 + r + hi * 8) * 64 + col] = c[r];
    } else {
#pragma unroll
        for (int r = 0; r < 8; ++r) {
            int orow = tm * 16 + r + hi * 8;
            if (orow < N) h2[(size_t)orow * 64 + col] = c[r];
        }
    }
}

// ------------------------------------------------- edge scatter: one wave/edge
// F=128: each lane owns 4 consecutive feats (b128 load); F=64: 2 feats (b64).
template<int F>
__global__ void __launch_bounds__(256) k_scatter(const int* __restrict__ src,
                                                 const int* __restrict__ dst,
                                                 const float* __restrict__ dis,
                                                 const float* __restrict__ h,
                                                 float* __restrict__ agg, int E) {
    int wid  = (blockIdx.x * blockDim.x + threadIdx.x) >> 5;
    int lane = threadIdx.x & 31;
    if (wid >= E) return;
    int s = src[wid], d = dst[wid];
    float w = dis[s] * dis[d];
    const float* __restrict__ hs = h + (size_t)s * F;
    float* __restrict__ ad = agg + (size_t)d * F;
    if constexpr (F == 128) {
        int f = lane * 4;
        float4 v = *(const float4*)(hs + f);
        atomicAdd(&ad[f + 0], w * v.x);
        atomicAdd(&ad[f + 1], w * v.y);
        atomicAdd(&ad[f + 2], w * v.z);
        atomicAdd(&ad[f + 3], w * v.w);
    } else {
        int f = lane * 2;
        float2 v = *(const float2*)(hs + f);
        atomicAdd(&ad[f + 0], w * v.x);
        atomicAdd(&ad[f + 1], w * v.y);
    }
}

// ------------------------------------------------- finalize layer1 (in place)
__global__ void k_fin1(float* __restrict__ agg1, const float* __restrict__ h1,
                       const float* __restrict__ dis, const float* __restrict__ b1,
                       int total4) {
    int i = blockIdx.x * blockDim.x + threadIdx.x;
    if (i >= total4) return;
    int i4 = i * 4;
    int n = i4 >> 7, f = i4 & 127;
    float dn = dis[n]; dn *= dn;
    float4 ag = *(const float4*)(agg1 + i4);
    float4 hh = *(const float4*)(h1 + i4);
    float4 bb = *(const float4*)(b1 + f);
    float4 o;
    o.x = fmaxf(ag.x + dn * hh.x + bb.x, 0.0f);
    o.y = fmaxf(ag.y + dn * hh.y + bb.y, 0.0f);
    o.z = fmaxf(ag.z + dn * hh.z + bb.z, 0.0f);
    o.w = fmaxf(ag.w + dn * hh.w + bb.w, 0.0f);
    *(float4*)(agg1 + i4) = o;
}

// ------------------------------------------------- finalize layer2 + pool
__global__ void k_fin2(const float* __restrict__ agg2, const float* __restrict__ h2,
                       const float* __restrict__ dis, const float* __restrict__ b2,
                       const int* __restrict__ batch, float* __restrict__ pooled,
                       int total4) {
    int i = blockIdx.x * blockDim.x + threadIdx.x;
    if (i >= total4) return;
    int i4 = i * 4;
    int n = i4 >> 6, f = i4 & 63;
    float dn = dis[n]; dn *= dn;
    float4 ag = *(const float4*)(agg2 + i4);
    float4 hh = *(const float4*)(h2 + i4);
    float4 bb = *(const float4*)(b2 + f);
    float* __restrict__ pg = pooled + (size_t)batch[n] * 64 + f;
    atomicAdd(pg + 0, fmaxf(ag.x + dn * hh.x + bb.x, 0.0f));
    atomicAdd(pg + 1, fmaxf(ag.y + dn * hh.y + bb.y, 0.0f));
    atomicAdd(pg + 2, fmaxf(ag.z + dn * hh.z + bb.z, 0.0f));
    atomicAdd(pg + 3, fmaxf(ag.w + dn * hh.w + bb.w, 0.0f));
}

__global__ void k_count(const int* __restrict__ batch, float* __restrict__ counts, int N) {
    int i = blockIdx.x * blockDim.x + threadIdx.x;
    if (i < N) atomicAdd(&counts[batch[i]], 1.0f);
}

// ------------------------------------------------- readout: wave per graph
__global__ void __launch_bounds__(256) k_readout(const float* __restrict__ pooled,
                                                 const float* __restrict__ counts,
                                                 const float* __restrict__ fcw,
                                                 const float* __restrict__ fcb,
                                                 float* __restrict__ out, int Gn) {
    int lane = threadIdx.x & 31;
    int g = (blockIdx.x * blockDim.x + threadIdx.x) >> 5;
    if (g >= Gn) return;
    float inv = 1.0f / fmaxf(counts[g], 1.0f);
    float acc = pooled[(size_t)g * 64 + lane] * inv * fcw[lane]
              + pooled[(size_t)g * 64 + 32 + lane] * inv * fcw[32 + lane];
    for (int off = 16; off > 0; off >>= 1)
        acc += __shfl_down(acc, off, 32);
    if (lane == 0)
        out[g] = 1.0f / (1.0f + __expf(-(acc + fcb[0])));
}

// ---------------------------------------------------------------- launcher
extern "C" void kernel_launch(void* const* d_in, const int* in_sizes, int n_in,
                              void* d_out, int out_size, void* d_ws, size_t ws_size,
                              hipStream_t stream) {
    const int*   node_ids = (const int*)d_in[0];
    const int*   edge_idx = (const int*)d_in[1];
    const int*   batch    = (const int*)d_in[2];
    const float* emb      = (const float*)d_in[3];
    const float* W1       = (const float*)d_in[4];
    const float* b1       = (const float*)d_in[5];
    const float* W2       = (const float*)d_in[6];
    const float* b2       = (const float*)d_in[7];
    const float* fcw      = (const float*)d_in[8];
    const float* fcb      = (const float*)d_in[9];
    float*       out      = (float*)d_out;

    const int N = in_sizes[0];
    const int E = in_sizes[1] / 2;
    const int G = out_size;
    const int* src = edge_idx;
    const int* dst = edge_idx + E;

    float* ws = (float*)d_ws;
    size_t o = 0;
    float* deg    = ws + o; o += (size_t)N;
    float* dis    = ws + o; o += (size_t)N;
    float* h1     = ws + o; o += (size_t)N * 128;   // reused: h2=[0,N*64), agg2=[N*64,N*128)
    float* agg1   = ws + o; o += (size_t)N * 128;   // becomes x1 after fin1
    float* pooled = ws + o; o += (size_t)G * 64;
    float* counts = ws + o; o += (size_t)G;
    _Float16* W1t = (_Float16*)(ws + o); o += (size_t)(64 * 128) / 2;
    _Float16* W2t = (_Float16*)(ws + o); o += (size_t)(128 * 64) / 2;
    float* h2   = h1;
    float* agg2 = h1 + (size_t)N * 64;
    (void)ws_size; (void)n_in; (void)o;

    const int T = 256;
    // init: deg=1 (self-loops), agg1=0, pooled+counts=0; pre-transpose weights
    k_fill<<<(N / 4 + T - 1) / T, T, 0, stream>>>(deg, 1.0f, N / 4);
    k_fill<<<(N * 32 + T - 1) / T, T, 0, stream>>>(agg1, 0.0f, N * 32);
    k_fill<<<(G * 65 / 4 + T - 1) / T, T, 0, stream>>>(pooled, 0.0f, G * 65 / 4);
    k_prepw<<<(64 * 128 + T - 1) / T, T, 0, stream>>>(W1, W1t, 64, 128);
    k_prepw<<<(128 * 64 + T - 1) / T, T, 0, stream>>>(W2, W2t, 128, 64);
    // degrees and symmetric norm factors
    k_degree<<<(E + T - 1) / T, T, 0, stream>>>(dst, deg, E);
    k_dis<<<(N + T - 1) / T, T, 0, stream>>>(deg, dis, N);
    // layer 1
    k_gemm1<<<(N + 15) / 16, T, 0, stream>>>(node_ids, emb, W1t, h1, N);
    k_scatter<128><<<(int)(((size_t)E * 32 + T - 1) / T), T, 0, stream>>>(src, dst, dis, h1, agg1, E);
    k_fin1<<<(N * 32 + T - 1) / T, T, 0, stream>>>(agg1, h1, dis, b1, N * 32);
    // layer 2 (h2 overwrites h1's low half; agg2 uses its high half)
    k_gemm2<<<((N + 15) / 16 + 1) / 2, T, 0, stream>>>(agg1, W2t, h2, N);
    k_fill<<<(N * 16 + T - 1) / T, T, 0, stream>>>(agg2, 0.0f, N * 16);
    k_scatter<64><<<(int)(((size_t)E * 32 + T - 1) / T), T, 0, stream>>>(src, dst, dis, h2, agg2, E);
    k_fin2<<<(N * 16 + T - 1) / T, T, 0, stream>>>(agg2, h2, dis, b2, batch, pooled, N * 16);
    k_count<<<(N + T - 1) / T, T, 0, stream>>>(batch, counts, N);
    // readout
    k_readout<<<(G * 32 + T - 1) / T, T, 0, stream>>>(pooled, counts, fcw, fcb, out, G);
}
